// VQVAE_89601607729465
// MI455X (gfx1250) — matrix operand order repaired and checked by reading, not compiled, not run on hardware
//
#include <hip/hip_runtime.h>

typedef __attribute__((ext_vector_type(16))) _Float16 v16h;
typedef __attribute__((ext_vector_type(8)))  _Float16 v8h;
typedef __attribute__((ext_vector_type(8)))  float    v8f;
typedef __attribute__((ext_vector_type(4)))  unsigned int u32x4;
typedef __attribute__((ext_vector_type(8)))  int      i32x8;
typedef __attribute__((ext_vector_type(4)))  int      i32x4;

#define CT_BLOCK 128
#define TSTR 72   // LDS tile row stride in halves: 64 data + 8 pad (bank-conflict-free)

#if __has_builtin(__builtin_amdgcn_tensor_load_to_lds)
#define HAVE_TDM 1
#else
#define HAVE_TDM 0
#endif

// ---------------------------------------------------------------------------
// TDM: 2D box tile (tile1 rows x tile0 elems of f16) from row-major global
// array into LDS, padded 4 DWORDs after every 32 DWORDs (row stride 72 halves).
// Descriptor packing per CDNA5 ISA ch.8 (group0: count/lds/global/type,
// group1: data_size=2B, pad_enable, dims/strides). Issued per-wave, EXEC
// ignored; completion via TENSORcnt.
// ---------------------------------------------------------------------------
__device__ __forceinline__ void tdm_load_tile_f16(
    const _Float16* gptr, unsigned ldsOff, int rowLen, int rows, int strideElems)
{
#if HAVE_TDM
    unsigned long long ga = (unsigned long long)(size_t)gptr;
    u32x4 g0;
    g0[0] = 1u;                                   // count=1, user mode
    g0[1] = ldsOff;                               // lds_addr (bytes)
    g0[2] = (unsigned)ga;                         // global_addr[31:0]
    g0[3] = (unsigned)(ga >> 32) | 0x80000000u;   // global_addr[56:32] | type=2
    i32x8 g1;
    g1[0] = (1 << 16)        // data_size = 2 bytes
          | (1 << 20)        // pad_enable
          | (4 << 22)        // pad_interval: 32 DWORDs (one 64-half row)
          | (3 << 25);       // pad_amount: 4 DWORDs (8 halves)
    g1[1] = (rowLen & 0xFFFF) << 16;                         // tensor_dim0 lo
    g1[2] = ((unsigned)rowLen >> 16) | ((rows & 0xFFFF) << 16); // dim0 hi | dim1 lo
    g1[3] = ((unsigned)rows >> 16) | (64 << 16);             // dim1 hi | tile_dim0=64
    g1[4] = 64;                                              // tile_dim1=64
    g1[5] = strideElems;                                     // tensor_dim0_stride lo
    g1[6] = 0;
    g1[7] = 0;
    i32x4 gz = (i32x4){0, 0, 0, 0};
#if defined(__clang_major__) && __clang_major__ >= 23
    i32x8 gz8 = (i32x8){0, 0, 0, 0, 0, 0, 0, 0};
    __builtin_amdgcn_tensor_load_to_lds(g0, g1, gz, gz, gz8, 0);
#else
    __builtin_amdgcn_tensor_load_to_lds(g0, g1, gz, gz, 0);
#endif
#endif
}

// Fallback cooperative weight-tile copy (only used if TDM builtin is absent).
__device__ __forceinline__ void coop_load_tile_f16(
    _Float16* Ws, const _Float16* gptr, int rowLen, int rows, int strideElems,
    int tid)
{
    for (int idx = tid; idx < 64 * 64; idx += CT_BLOCK) {
        int rr = idx >> 6, kk = idx & 63;
        _Float16 v = (_Float16)0.0f;
        if (rr < rows && kk < rowLen) v = gptr[rr * strideElems + kk];
        Ws[rr * TSTR + kk] = v;
    }
}

// ---------------------------------------------------------------------------
// Implicit-GEMM conv (k=4,s=2,p=1), WMMA f16->f32, K-step 64.
//   A = weights [Cout][K] (TDM-loaded box tile), B = im2col pixels [m][k],
//   D rows = Cout, cols = pixels (stores contiguous along W).
//   K = Cin*16, square images, Wout = 1<<woShift, Hin = Win = 2*Wout.
// ---------------------------------------------------------------------------
__global__ __launch_bounds__(CT_BLOCK) void conv_wmma_kernel(
    const _Float16* __restrict__ in, const _Float16* __restrict__ w16,
    const float* __restrict__ bias, _Float16* __restrict__ out16,
    float* __restrict__ out32,
    int Cin, int Hin, int Cout, int woShift, int M, int Kreal, int relu)
{
    __shared__ __align__(16) _Float16 Ws[64 * TSTR];  // weights [co][k]
    __shared__ __align__(16) _Float16 Ps[64 * TSTR];  // pixels  [m][k]

    const int tid     = threadIdx.x;
    const int lane    = tid & 31;
    const int wv      = tid >> 5;
    const int pixBase = blockIdx.x * 64;
    const int coBase  = blockIdx.y * 64;
    const int Wout    = 1 << woShift;
    const int Win     = Hin;
    const int hwMask  = (1 << (2 * woShift)) - 1;
    const int wMask   = Wout - 1;
    const unsigned wsOff = (unsigned)(unsigned long long)(size_t)&Ws[0];

    v8f acc[4];
    #pragma unroll
    for (int j = 0; j < 4; ++j) acc[j] = (v8f){0.f,0.f,0.f,0.f,0.f,0.f,0.f,0.f};

    const int Ksteps = (Kreal + 63) >> 6;
    for (int ks = 0; ks < Ksteps; ++ks) {
        const int k0 = ks << 6;
#if HAVE_TDM
        if (wv == 0)
            tdm_load_tile_f16(w16 + (size_t)coBase * Kreal + k0, wsOff,
                              Kreal - k0, Cout - coBase, Kreal);
#else
        coop_load_tile_f16(Ws, w16 + (size_t)coBase * Kreal + k0,
                           Kreal - k0, Cout - coBase, Kreal, tid);
#endif
        // stage im2col pixel tile 64 m-rows x 64 k (overlaps the TDM DMA)
        for (int idx = tid; idx < 64 * 64; idx += CT_BLOCK) {
            int ml = idx >> 6, kl = idx & 63;
            int m = pixBase + ml, k = k0 + kl;
            _Float16 v = (_Float16)0.0f;
            if (m < M && k < Kreal) {
                int n  = m >> (2 * woShift);
                int hw = m & hwMask;
                int ho = hw >> woShift, wo = hw & wMask;
                int ci = k >> 4, r = k & 15, kh = r >> 2, kw = r & 3;
                int hi = (ho << 1) - 1 + kh;
                int wi = (wo << 1) - 1 + kw;
                if ((unsigned)hi < (unsigned)Hin && (unsigned)wi < (unsigned)Win)
                    v = in[((n * Cin + ci) * Hin + hi) * Win + wi];
            }
            Ps[ml * TSTR + kl] = v;
        }
#if HAVE_TDM
        if (wv == 0) __builtin_amdgcn_s_wait_tensorcnt(0);
#endif
        __syncthreads();

        // 2 K-substeps x 4 pixel groups = 8 WMMAs per wave per stage
        const int arow = (wv << 4) + (lane & 15);
        const int ahalf = (lane & 16) ? 8 : 0;
        const int bhalf = (lane & 16) ? 8 : 0;
        #pragma unroll
        for (int s = 0; s < 2; ++s) {
            const int akb = (s << 5) + ahalf;
            v8h a0 = *(const v8h*)&Ws[arow * TSTR + akb];
            v8h a1 = *(const v8h*)&Ws[arow * TSTR + akb + 16];
            v16h afrag = __builtin_shufflevector(a0, a1,
                            0,1,2,3,4,5,6,7,8,9,10,11,12,13,14,15);
            const int bkb = (s << 5) + bhalf;
            #pragma unroll
            for (int j = 0; j < 4; ++j) {
                int brow = (j << 4) + (lane & 15);
                v8h b0 = *(const v8h*)&Ps[brow * TSTR + bkb];
                v8h b1 = *(const v8h*)&Ps[brow * TSTR + bkb + 16];
                v16h bfrag = __builtin_shufflevector(b0, b1,
                                0,1,2,3,4,5,6,7,8,9,10,11,12,13,14,15);
                acc[j] = __builtin_amdgcn_wmma_f32_16x16x32_f16(
                             false, afrag, false, bfrag, (short)0, acc[j],
                             false, false);
            }
        }
        __syncthreads();
    }

    // D: lane holds pixel N = lane&15, cout M = r + 8*(lane>=16), wave = co/16
    const int cl    = lane & 15;
    const int coRow = coBase + (wv << 4) + ((lane & 16) ? 8 : 0);
    float bb[8];
    #pragma unroll
    for (int r = 0; r < 8; ++r) {
        int co = coRow + r;
        bb[r] = (co < Cout) ? bias[co] : 0.f;
    }
    #pragma unroll
    for (int j = 0; j < 4; ++j) {
        int m = pixBase + (j << 4) + cl;
        if (m >= M) continue;
        int n  = m >> (2 * woShift);
        int hw = m & hwMask;
        int ho = hw >> woShift, wo = hw & wMask;
        int obase = (n * Cout * Wout + ho) * Wout + wo;
        #pragma unroll
        for (int r = 0; r < 8; ++r) {
            int co = coRow + r;
            if (co >= Cout) continue;
            float v = acc[j][r] + bb[r];
            if (relu) v = v > 0.f ? v : 0.f;
            int o = obase + co * Wout * Wout;
            if (out16) out16[o] = (_Float16)v;
            if (out32) out32[o] = v;
        }
    }
}

// ---------------------------------------------------------------------------
// Transposed conv (k=4,s=2,p=1) as 4 parity sub-convs with 2x2 taps.
//   blockIdx.z = parity p = ph*2+pw; per parity M = N*Hin*Win, K = Cin*4,
//   weights pre-transformed to wT[p][Cout][K] (TDM box tile).
//   ph==0: th0->(kh=1,dh=0),  th1->(kh=3,dh=-1)
//   ph==1: th0->(kh=0,dh=+1), th1->(kh=2,dh=0)   (same for w)
// ---------------------------------------------------------------------------
__global__ __launch_bounds__(CT_BLOCK) void deconv_wmma_kernel(
    const _Float16* __restrict__ in, const _Float16* __restrict__ wT,
    const float* __restrict__ bias, _Float16* __restrict__ out16,
    float* __restrict__ out32,
    int Cin, int Hin, int Cout, int winShift, int M, int relu)
{
    __shared__ __align__(16) _Float16 Ws[64 * TSTR];
    __shared__ __align__(16) _Float16 Ps[64 * TSTR];

    const int tid     = threadIdx.x;
    const int lane    = tid & 31;
    const int wv      = tid >> 5;
    const int pixBase = blockIdx.x * 64;
    const int coBase  = blockIdx.y * 64;
    const int p       = blockIdx.z;
    const int ph      = p >> 1, pw = p & 1;
    const int Win     = 1 << winShift;
    const int hwMask  = (1 << (2 * winShift)) - 1;
    const int wMask   = Win - 1;
    const int Kreal   = Cin * 4;
    const _Float16* wTp = wT + (size_t)p * Cout * Kreal;
    const unsigned wsOff = (unsigned)(unsigned long long)(size_t)&Ws[0];

    v8f acc[4];
    #pragma unroll
    for (int j = 0; j < 4; ++j) acc[j] = (v8f){0.f,0.f,0.f,0.f,0.f,0.f,0.f,0.f};

    const int Ksteps = (Kreal + 63) >> 6;
    for (int ks = 0; ks < Ksteps; ++ks) {
        const int k0 = ks << 6;
#if HAVE_TDM
        if (wv == 0)
            tdm_load_tile_f16(wTp + (size_t)coBase * Kreal + k0, wsOff,
                              Kreal - k0, Cout - coBase, Kreal);
#else
        coop_load_tile_f16(Ws, wTp + (size_t)coBase * Kreal + k0,
                           Kreal - k0, Cout - coBase, Kreal, tid);
#endif
        for (int idx = tid; idx < 64 * 64; idx += CT_BLOCK) {
            int ml = idx >> 6, kl = idx & 63;
            int m = pixBase + ml, k = k0 + kl;
            _Float16 v = (_Float16)0.0f;
            if (m < M && k < Kreal) {
                int n  = m >> (2 * winShift);
                int hw = m & hwMask;
                int hb = hw >> winShift, wb = hw & wMask;
                int ci = k >> 2, t = k & 3, th = t >> 1, tw = t & 1;
                int hi = hb + (ph ? (th ? 0 : 1) : (th ? -1 : 0));
                int wi = wb + (pw ? (tw ? 0 : 1) : (tw ? -1 : 0));
                if ((unsigned)hi < (unsigned)Hin && (unsigned)wi < (unsigned)Win)
                    v = in[((n * Cin + ci) * Hin + hi) * Win + wi];
            }
            Ps[ml * TSTR + kl] = v;
        }
#if HAVE_TDM
        if (wv == 0) __builtin_amdgcn_s_wait_tensorcnt(0);
#endif
        __syncthreads();

        const int arow = (wv << 4) + (lane & 15);
        const int ahalf = (lane & 16) ? 8 : 0;
        const int bhalf = (lane & 16) ? 8 : 0;
        #pragma unroll
        for (int s = 0; s < 2; ++s) {
            const int akb = (s << 5) + ahalf;
            v8h a0 = *(const v8h*)&Ws[arow * TSTR + akb];
            v8h a1 = *(const v8h*)&Ws[arow * TSTR + akb + 16];
            v16h afrag = __builtin_shufflevector(a0, a1,
                            0,1,2,3,4,5,6,7,8,9,10,11,12,13,14,15);
            const int bkb = (s << 5) + bhalf;
            #pragma unroll
            for (int j = 0; j < 4; ++j) {
                int brow = (j << 4) + (lane & 15);
                v8h b0 = *(const v8h*)&Ps[brow * TSTR + bkb];
                v8h b1 = *(const v8h*)&Ps[brow * TSTR + bkb + 16];
                v16h bfrag = __builtin_shufflevector(b0, b1,
                                0,1,2,3,4,5,6,7,8,9,10,11,12,13,14,15);
                acc[j] = __builtin_amdgcn_wmma_f32_16x16x32_f16(
                             false, afrag, false, bfrag, (short)0, acc[j],
                             false, false);
            }
        }
        __syncthreads();
    }

    const int cl    = lane & 15;
    const int coRow = coBase + (wv << 4) + ((lane & 16) ? 8 : 0);
    const int Hout  = Hin * 2, Wout = Win * 2;
    float bb[8];
    #pragma unroll
    for (int r = 0; r < 8; ++r) {
        int co = coRow + r;
        bb[r] = (co < Cout) ? bias[co] : 0.f;
    }
    #pragma unroll
    for (int j = 0; j < 4; ++j) {
        int m = pixBase + (j << 4) + cl;
        if (m >= M) continue;
        int n  = m >> (2 * winShift);
        int hw = m & hwMask;
        int hb = hw >> winShift, wb = hw & wMask;
        int ho = (hb << 1) + ph, wo = (wb << 1) + pw;
        int obase = (n * Cout * Hout + ho) * Wout + wo;
        #pragma unroll
        for (int r = 0; r < 8; ++r) {
            int co = coRow + r;
            if (co >= Cout) continue;
            float v = acc[j][r] + bb[r];
            if (relu) v = v > 0.f ? v : 0.f;
            int o = obase + co * Hout * Wout;
            if (out16) out16[o] = (_Float16)v;
            if (out32) out32[o] = v;
        }
    }
}

// ---------------------------------------------------------------------------
// Vector quantization: per row (64-dim) argmin over 512 codes.
// Ranking needs only |e|^2 - 2 x.e (|x|^2 constant per row).
// ---------------------------------------------------------------------------
__global__ __launch_bounds__(256) void vq_kernel(
    const float* __restrict__ z_e, const float* __restrict__ emb,
    const float* __restrict__ enorm, float* __restrict__ zq_out,
    float* __restrict__ idx_out, _Float16* __restrict__ zq16, int nrows)
{
    int m = blockIdx.x * blockDim.x + threadIdx.x;
    if (m >= nrows) return;
    float x[64];
    #pragma unroll
    for (int i = 0; i < 64; ++i) x[i] = z_e[m * 64 + i];
    float best = 3.4e38f;
    int bi = 0;
    for (int e = 0; e < 512; ++e) {
        float dot = 0.f;
        #pragma unroll
        for (int i = 0; i < 64; ++i) dot = fmaf(x[i], emb[e * 64 + i], dot);
        float d = enorm[e] - 2.f * dot;
        if (d < best) { best = d; bi = e; }
    }
    idx_out[m] = (float)bi;
    #pragma unroll
    for (int i = 0; i < 64; ++i) {
        float v = emb[bi * 64 + i];
        zq_out[m * 64 + i] = v;
        zq16[m * 64 + i]   = (_Float16)v;
    }
}

// ---------------------------------------------------------------------------
// Prep kernels.
// ---------------------------------------------------------------------------
__global__ void cast_f16_kernel(const float* __restrict__ src,
                                _Float16* __restrict__ dst, int n) {
    int i = blockIdx.x * blockDim.x + threadIdx.x;
    if (i < n) dst[i] = (_Float16)src[i];
}

// deconv weights [Cin][Cout][4][4] -> dT[p][Cout][K], K = Cin*4
__global__ void prep_deconv_w_kernel(const float* __restrict__ w,
                                     _Float16* __restrict__ dT, int Cin, int Cout) {
    int K = Cin * 4;
    int total = 4 * Cout * K;
    int i = blockIdx.x * blockDim.x + threadIdx.x;
    if (i < total) {
        int k  = i % K;
        int r  = i / K;
        int co = r % Cout;
        int p  = r / Cout;
        int ci = k >> 2, t = k & 3;
        int ph = p >> 1, pw = p & 1;
        int th = t >> 1, tw = t & 1;
        int kh = ph ? (th ? 2 : 0) : (th ? 3 : 1);
        int kw = pw ? (tw ? 2 : 0) : (tw ? 3 : 1);
        dT[i] = (_Float16)w[((ci * Cout + co) * 4 + kh) * 4 + kw];
    }
}

__global__ void prep_enorm_kernel(const float* __restrict__ emb,
                                  float* __restrict__ enorm) {
    int e = blockIdx.x * blockDim.x + threadIdx.x;
    if (e < 512) {
        float s = 0.f;
        #pragma unroll
        for (int i = 0; i < 64; ++i) {
            float v = emb[e * 64 + i];
            s = fmaf(v, v, s);
        }
        enorm[e] = s;
    }
}

// ---------------------------------------------------------------------------
extern "C" void kernel_launch(void* const* d_in, const int* in_sizes, int n_in,
                              void* d_out, int out_size, void* d_ws, size_t ws_size,
                              hipStream_t stream) {
    const float* x   = (const float*)d_in[0];
    const float* w1  = (const float*)d_in[1];
    const float* b1  = (const float*)d_in[2];
    const float* w2  = (const float*)d_in[3];
    const float* b2  = (const float*)d_in[4];
    const float* w3  = (const float*)d_in[5];
    const float* b3  = (const float*)d_in[6];
    const float* d1w = (const float*)d_in[7];
    const float* d1b = (const float*)d_in[8];
    const float* d2w = (const float*)d_in[9];
    const float* d2b = (const float*)d_in[10];
    const float* d3w = (const float*)d_in[11];
    const float* d3b = (const float*)d_in[12];
    const float* emb = (const float*)d_in[13];

    float* out = (float*)d_out;
    const int XR = 16 * 3 * 256 * 256;  // 3145728
    const int ZE = 16 * 64 * 32 * 32;   // 1048576
    float* xrec_o = out;
    float* ze_o   = out + XR;
    float* zq_o   = out + XR + ZE;
    float* idx_o  = out + XR + 2 * ZE;

    char*  ws  = (char*)d_ws;
    size_t off = 0;
    auto alloc = [&](size_t bytes) -> void* {
        void* pp = ws + off;
        off = (off + bytes + 255) & ~(size_t)255;
        return pp;
    };
    _Float16* x16  = (_Float16*)alloc((size_t)XR * 2);              // 6 MiB
    _Float16* h1   = (_Float16*)alloc((size_t)16*128*128*128 * 2);  // 64 MiB (reused as g2)
    _Float16* h2   = (_Float16*)alloc((size_t)16*256*64*64 * 2);    // 32 MiB (reused as g1)
    _Float16* zq16 = (_Float16*)alloc((size_t)ZE * 2);              // 2 MiB
    _Float16* w116 = (_Float16*)alloc((size_t)128 * 48   * 2);      // [Cout][K] f16
    _Float16* w216 = (_Float16*)alloc((size_t)256 * 2048 * 2);
    _Float16* w316 = (_Float16*)alloc((size_t)64  * 4096 * 2);
    _Float16* d1T  = (_Float16*)alloc((size_t)4 * 256 * 256  * 2);  // [p][Cout][K]
    _Float16* d2T  = (_Float16*)alloc((size_t)4 * 128 * 1024 * 2);
    _Float16* d3T  = (_Float16*)alloc((size_t)4 * 3   * 512  * 2);
    float*    enorm= (float*)   alloc(512 * 4);
    _Float16* g1 = h2;
    _Float16* g2 = h1;

    // ---- prep ----
    cast_f16_kernel<<<(XR + 255) / 256, 256, 0, stream>>>(x, x16, XR);
    cast_f16_kernel<<<(128*48   + 255)/256, 256, 0, stream>>>(w1, w116, 128*48);
    cast_f16_kernel<<<(256*2048 + 255)/256, 256, 0, stream>>>(w2, w216, 256*2048);
    cast_f16_kernel<<<(64*4096  + 255)/256, 256, 0, stream>>>(w3, w316, 64*4096);
    prep_deconv_w_kernel<<<(4*256*256  + 255)/256, 256, 0, stream>>>(d1w, d1T, 64,  256);
    prep_deconv_w_kernel<<<(4*128*1024 + 255)/256, 256, 0, stream>>>(d2w, d2T, 256, 128);
    prep_deconv_w_kernel<<<(4*3*512    + 255)/256, 256, 0, stream>>>(d3w, d3T, 128, 3);
    prep_enorm_kernel<<<2, 256, 0, stream>>>(emb, enorm);

    // ---- encoder ----
    // conv1: 3->128, 256->128, M=262144, K=48
    conv_wmma_kernel<<<dim3(262144/64, 2), CT_BLOCK, 0, stream>>>(
        x16, w116, b1, h1, nullptr, 3, 256, 128, 7, 262144, 48, 1);
    // conv2: 128->256, 128->64, M=65536, K=2048
    conv_wmma_kernel<<<dim3(65536/64, 4), CT_BLOCK, 0, stream>>>(
        h1, w216, b2, h2, nullptr, 128, 128, 256, 6, 65536, 2048, 1);
    // conv3: 256->64, 64->32, M=16384, K=4096 -> z_e (fp32, to d_out)
    conv_wmma_kernel<<<dim3(16384/64, 1), CT_BLOCK, 0, stream>>>(
        h2, w316, b3, nullptr, ze_o, 256, 64, 64, 5, 16384, 4096, 0);

    // ---- quantize ----
    vq_kernel<<<16384 / 256, 256, 0, stream>>>(ze_o, emb, enorm, zq_o, idx_o,
                                               zq16, 16384);

    // ---- decoder (per-parity M, grid.z = parity) ----
    // deconv1: 64->256, 32->64, Mp=16384, K=256
    deconv_wmma_kernel<<<dim3(16384/64, 4, 4), CT_BLOCK, 0, stream>>>(
        zq16, d1T, d1b, g1, nullptr, 64, 32, 256, 5, 16384, 1);
    // deconv2: 256->128, 64->128, Mp=65536, K=1024
    deconv_wmma_kernel<<<dim3(65536/64, 2, 4), CT_BLOCK, 0, stream>>>(
        g1, d2T, d2b, g2, nullptr, 256, 64, 128, 6, 65536, 1);
    // deconv3: 128->3, 128->256, Mp=262144, K=512 -> x_recon (fp32, to d_out)
    deconv_wmma_kernel<<<dim3(262144/64, 1, 4), CT_BLOCK, 0, stream>>>(
        g2, d3T, d3b, nullptr, xrec_o, 128, 128, 3, 7, 262144, 0);
}
